// StackRNN_26645977104997
// MI455X (gfx1250) — matrix operand order repaired and checked by reading, not compiled, not run on hardware
//
#include <hip/hip_runtime.h>
#include <hip/hip_bf16.h>
#include <cstdint>
#include <cstddef>

// ---------------------------------------------------------------------------
// MI455X (gfx1250) implementation of the char+sentence stacked BiLSTM tagger.
// All GEMMs run through v_wmma_f32_16x16x32_bf16 (fp32 accumulate).
// Weights are packed ONCE per call into a B-fragment-native bf16 tile layout:
//   tile(ntile, kk) at offset (ntile*Ktiles + kk)*512 bf16; within a tile,
//   lane L owns 16 contiguous bf16 = K rows (L>=16 ? 16..31 : 0..15) of
//   column N = ntile*16 + (L&15)  -> one coalesced 32B load per lane.
// ---------------------------------------------------------------------------

typedef unsigned short u16;
typedef __attribute__((ext_vector_type(16))) __bf16   v16bf;
typedef __attribute__((ext_vector_type(8)))  float    v8f;
typedef __attribute__((ext_vector_type(4)))  unsigned uv4;

union Frag {
  uv4      q[2];
  unsigned u[8];
  v16bf    v;
};

__device__ __forceinline__ u16 f2bf(float x) {
  union { float f; unsigned u; } c; c.f = x;
  unsigned r = c.u + 0x7FFFu + ((c.u >> 16) & 1u);   // round-to-nearest-even
  return (u16)(r >> 16);
}
__device__ __forceinline__ float sigf(float x) { return 1.f / (1.f + __expf(-x)); }

// ---------------------------------------------------------------- utilities
__global__ void k_zero_f32(float* __restrict__ p, int n) {
  int i = blockIdx.x * 256 + threadIdx.x;
  if (i < n) p[i] = 0.f;
}

// Pack combined LSTM weight [Wx;Wh] (fp32, row-major [K][4H]) into B tiles.
// K layout: [0,kxpad) = x rows (kxv valid), [kxpad, ...) = h rows (khv valid).
// N layout: gate*Hpad + j (j<H valid), zero padding elsewhere.
__global__ void k_pack_lstm_w(const float* __restrict__ Wx, const float* __restrict__ Wh,
                              u16* __restrict__ dst, int ktiles, int kxpad, int kxv, int khv,
                              int H, int Hpad, int total) {
  int idx = blockIdx.x * 256 + threadIdx.x;
  if (idx >= total) return;
  int tile = idx >> 9, p = idx & 511;
  int lane = p >> 4, j16 = p & 15;
  int kk = tile % ktiles, nt = tile / ktiles;
  int k = kk * 32 + (lane >> 4) * 16 + j16;
  int n = nt * 16 + (lane & 15);
  int gate = n / Hpad, j = n % Hpad;
  float v = 0.f;
  if (j < H) {
    if (k < kxpad) { if (k < kxv) v = Wx[(size_t)k * (4 * H) + gate * H + j]; }
    else           { int r = k - kxpad; if (r < khv) v = Wh[(size_t)r * (4 * H) + gate * H + j]; }
  }
  dst[idx] = f2bf(v);
}

// Pack dense-head weight [Kvalid][Nvalid] into B tiles (zero padded).
__global__ void k_pack_head_w(const float* __restrict__ src, u16* __restrict__ dst,
                              int ktiles, int kvalid, int nvalid, int total) {
  int idx = blockIdx.x * 256 + threadIdx.x;
  if (idx >= total) return;
  int tile = idx >> 9, p = idx & 511;
  int lane = p >> 4, j16 = p & 15;
  int kk = tile % ktiles, nt = tile / ktiles;
  int k = kk * 32 + (lane >> 4) * 16 + j16;
  int n = nt * 16 + (lane & 15);
  float v = (k < kvalid && n < nvalid) ? src[(size_t)k * nvalid + n] : 0.f;
  dst[idx] = f2bf(v);
}

// Pack LSTM bias with TF forget_bias=1.0 folded into gate 2 (i,g,f,o order).
__global__ void k_pack_lstm_b(const float* __restrict__ b, float* __restrict__ dst,
                              int H, int Hpad) {
  int idx = blockIdx.x * 256 + threadIdx.x;
  if (idx >= 4 * Hpad) return;
  int gate = idx / Hpad, j = idx % Hpad;
  dst[idx] = (j < H) ? (b[gate * H + j] + (gate == 2 ? 1.f : 0.f)) : 0.f;
}

// ------------------------------------------------------- char LSTM one step
// U=16384 rows, CE=100 (Hpad=112), K = 128(x pad) + 128(h pad) = 256.
// grid (1024 m-tiles, 7 j-tiles), 128 threads = 4 waves (one per gate).
#define CHP 112
#define CKP 264   // LDS row pitch (bf16) — avoids 64-bank conflicts
__global__ void __launch_bounds__(128)
k_char_step(const int* __restrict__ seqs, const int* __restrict__ lens,
            const float* __restrict__ emb,
            const float* __restrict__ hIn, float* __restrict__ hOut,
            float* __restrict__ Cst,
            const u16* __restrict__ Wp, const float* __restrict__ biasp,
            int t, int rev) {
  __shared__ u16  At[16 * CKP];
  __shared__ float gb[4][16][16];
  __shared__ int  ttr[16];
  int mt = blockIdx.x, jt = blockIdx.y;
  int m0 = mt * 16, j0 = jt * 16;
  int tid = threadIdx.x;

  if (tid < 16) {
    int L = lens[m0 + tid];
    int tt = rev ? (L - 1 - t) : t;
    ttr[tid] = tt < 0 ? 0 : (tt > 15 ? 15 : tt);
  }
  __syncthreads();

  // Stage A = [x_t | h_prev] in bf16. x_t = char_emb[charseqs[u, tt]].
  for (int l = tid; l < 16 * 256; l += 128) {
    int r = l >> 8, c = l & 255;
    int u = m0 + r;
    float v = 0.f;
    if (c < 128) {
      if (c < 100) { int ch = seqs[u * 16 + ttr[r]]; v = emb[ch * 100 + c]; }
    } else {
      int j = c - 128;
      if (j < 100) v = hIn[(size_t)u * CHP + j];
    }
    At[r * CKP + c] = f2bf(v);
  }
  __syncthreads();

  int lane = tid & 31, gate = tid >> 5;
  int row = lane & 15, khalf = (lane >> 4) * 8;
  const u16* wb = Wp + (size_t)(gate * 7 + jt) * 8 * 512 + lane * 16;
  v8f acc = {0.f, 0.f, 0.f, 0.f, 0.f, 0.f, 0.f, 0.f};
  for (int kk = 0; kk < 8; ++kk) {
    __builtin_prefetch(wb + (kk + 1 < 8 ? kk + 1 : kk) * 512, 0, 1);
    Frag a, b;
    const uv4* bp = (const uv4*)(wb + kk * 512);
    b.q[0] = bp[0]; b.q[1] = bp[1];
    int kb = kk * 32;
#pragma unroll
    for (int i = 0; i < 8; ++i) {
      int k = kb + 2 * khalf + 2 * i;
      a.u[i] = *(const unsigned*)&At[row * CKP + k];
    }
    acc = __builtin_amdgcn_wmma_f32_16x16x32_bf16(false, a.v, false, b.v, (short)0, acc, false, false);
  }
  {
    int n = lane & 15, mh = (lane >> 4) * 8;
#pragma unroll
    for (int i = 0; i < 8; ++i)
      gb[gate][mh + i][n] = acc[i] + biasp[gate * CHP + j0 + n];
  }
  __syncthreads();

  for (int e = tid; e < 256; e += 128) {
    int m = e >> 4, n = e & 15;
    int u = m0 + m, j = j0 + n;
    float gi = gb[0][m][n], gg = gb[1][m][n], gf = gb[2][m][n], go = gb[3][m][n];
    float co = Cst[(size_t)u * CHP + j];
    float ho = hIn[(size_t)u * CHP + j];
    float cn = sigf(gf) * co + sigf(gi) * tanhf(gg);
    float hn = tanhf(cn) * sigf(go);
    bool mk = t < lens[u];
    Cst[(size_t)u * CHP + j]  = mk ? cn : co;
    hOut[(size_t)u * CHP + j] = mk ? hn : ho;
  }
}

// --------------------------------------------------- sentence LSTM one step
// B=256 rows, H=500 (Hpad=512), K = 512(x) + 512(h) = 1024, S=128.
// grid (16 m-tiles, 32 j-tiles), 128 threads = 4 waves (one per gate).
#define SKP 1032  // LDS row pitch (bf16)
__global__ void __launch_bounds__(128)
k_sent_step(const u16* __restrict__ x, u16* __restrict__ outp,
            const float* __restrict__ hIn, float* __restrict__ hOut,
            float* __restrict__ Cst,
            const u16* __restrict__ Wp, const float* __restrict__ biasp,
            const int* __restrict__ lens, int t, int rev) {
  __shared__ u16  At[16 * SKP];
  __shared__ float gb[4][16][16];
  __shared__ int  ttr[16];
  int mt = blockIdx.x, jt = blockIdx.y;
  int m0 = mt * 16, j0 = jt * 16;
  int tid = threadIdx.x;

  if (tid < 16) {
    int L = lens[m0 + tid];
    int tt = rev ? (L - 1 - t) : t;
    ttr[tid] = tt < 0 ? 0 : (tt > 127 ? 127 : tt);
  }
  __syncthreads();

  for (int l = tid; l < 16 * 1024; l += 128) {
    int r = l >> 10, c = l & 1023;
    int b = m0 + r;
    if (c < 512) At[r * SKP + c] = x[((size_t)b * 128 + ttr[r]) * 512 + c];
    else         At[r * SKP + c] = f2bf(hIn[(size_t)b * 512 + (c - 512)]);
  }
  __syncthreads();

  int lane = tid & 31, gate = tid >> 5;
  int row = lane & 15, khalf = (lane >> 4) * 8;
  const u16* wb = Wp + (size_t)(gate * 32 + jt) * 32 * 512 + lane * 16;
  v8f acc = {0.f, 0.f, 0.f, 0.f, 0.f, 0.f, 0.f, 0.f};
  for (int kk = 0; kk < 32; ++kk) {
    __builtin_prefetch(wb + (kk + 1 < 32 ? kk + 1 : kk) * 512, 0, 1);
    Frag a, b;
    const uv4* bp = (const uv4*)(wb + kk * 512);
    b.q[0] = bp[0]; b.q[1] = bp[1];
    int kb = kk * 32;
#pragma unroll
    for (int i = 0; i < 8; ++i) {
      int k = kb + 2 * khalf + 2 * i;
      a.u[i] = *(const unsigned*)&At[row * SKP + k];
    }
    acc = __builtin_amdgcn_wmma_f32_16x16x32_bf16(false, a.v, false, b.v, (short)0, acc, false, false);
  }
  {
    int n = lane & 15, mh = (lane >> 4) * 8;
#pragma unroll
    for (int i = 0; i < 8; ++i)
      gb[gate][mh + i][n] = acc[i] + biasp[gate * 512 + j0 + n];
  }
  __syncthreads();

  for (int e = tid; e < 256; e += 128) {
    int m = e >> 4, n = e & 15;
    int b = m0 + m, j = j0 + n;
    float gi = gb[0][m][n], gg = gb[1][m][n], gf = gb[2][m][n], go = gb[3][m][n];
    float co = Cst[(size_t)b * 512 + j];
    float ho = hIn[(size_t)b * 512 + j];
    float cn = sigf(gf) * co + sigf(gi) * tanhf(gg);
    float hn = tanhf(cn) * sigf(go);
    bool mk = t < lens[b];
    Cst[(size_t)b * 512 + j]  = mk ? cn : co;
    hOut[(size_t)b * 512 + j] = mk ? hn : ho;
    if (outp) outp[((size_t)b * 128 + t) * 512 + j] = f2bf(mk ? hn : 0.f);
  }
}

// ------------------------- build sentence input: [char c/h gather | word emb]
__global__ void k_gather_inp(const int* __restrict__ cids, const int* __restrict__ wids,
                             const float* __restrict__ cf, const float* __restrict__ hf,
                             const float* __restrict__ cb, const float* __restrict__ hb,
                             const float* __restrict__ wemb, u16* __restrict__ dst) {
  int idx = blockIdx.x * 256 + threadIdx.x;
  if (idx >= 256 * 128 * 512) return;
  int bs = idx >> 9, c = idx & 511;
  float v = 0.f;
  if (c < 400) {
    int u = cids[bs];
    if      (c < 100) v = cf[(size_t)u * CHP + c];
    else if (c < 200) v = hf[(size_t)u * CHP + c - 100];
    else if (c < 300) v = cb[(size_t)u * CHP + c - 200];
    else              v = hb[(size_t)u * CHP + c - 300];
  } else if (c < 500) {
    int w = wids[bs];
    v = wemb[(size_t)w * 100 + (c - 400)];
  }
  dst[idx] = f2bf(v);
}

// --------------------- pack s = [c_f0 h_f0 c_f1 h_f1 c_b0 h_b0 c_b1 h_b1] (4000->4096)
__global__ void k_pack_s(u16* __restrict__ dst,
                         const float* c0, const float* c1, const float* c2, const float* c3,
                         const float* h0, const float* h1, const float* h2, const float* h3) {
  int idx = blockIdx.x * 256 + threadIdx.x;
  if (idx >= 256 * 4096) return;
  int b = idx >> 12, c = idx & 4095;
  float v = 0.f;
  if (c < 4000) {
    int chunk = c / 500, j = c % 500;
    const float* cs[8] = {c0, h0, c1, h1, c2, h2, c3, h3};
    v = cs[chunk][(size_t)b * 512 + j];
  }
  dst[idx] = f2bf(v);
}

// ----------------------------------------------------------- dense head GEMM
// A bf16 [M][kpitch] row-major, B packed tiles. One wave per 16x16 out tile.
__global__ void __launch_bounds__(128)
k_head_gemm(const u16* __restrict__ A, int kpitch, int ktiles,
            const u16* __restrict__ Bp, const float* __restrict__ bias, int nvalid,
            int dorelu, u16* __restrict__ outBf, float* __restrict__ outF,
            int ntiles, int npitch) {
  int mt = blockIdx.x;
  int nt = blockIdx.y * 4 + (threadIdx.x >> 5);
  if (nt >= ntiles) return;
  int lane = threadIdx.x & 31;
  int m0 = mt * 16;
  int row = lane & 15, khalf = (lane >> 4) * 8;
  const u16* arow = A + (size_t)(m0 + row) * kpitch;
  const u16* wb = Bp + (size_t)nt * ktiles * 512 + lane * 16;
  v8f acc = {0.f, 0.f, 0.f, 0.f, 0.f, 0.f, 0.f, 0.f};
  for (int kk = 0; kk < ktiles; ++kk) {
    __builtin_prefetch(wb + (kk + 1 < ktiles ? kk + 1 : kk) * 512, 0, 1);
    Frag a, b;
    const uv4* bp = (const uv4*)(wb + kk * 512);
    b.q[0] = bp[0]; b.q[1] = bp[1];
    int kb = kk * 32;
#pragma unroll
    for (int i = 0; i < 8; ++i) {
      int k = kb + 2 * khalf + 2 * i;
      a.u[i] = *(const unsigned*)&arow[k];
    }
    acc = __builtin_amdgcn_wmma_f32_16x16x32_bf16(false, a.v, false, b.v, (short)0, acc, false, false);
  }
  int n = lane & 15, mh = (lane >> 4) * 8;
#pragma unroll
  for (int i = 0; i < 8; ++i) {
    int ng = nt * 16 + n;
    float v = acc[i] + (ng < nvalid ? bias[ng] : 0.f);
    if (dorelu) v = v > 0.f ? v : 0.f;
    int m = m0 + mh + i;
    if (outBf) outBf[(size_t)m * npitch + ng] = f2bf(v);
    if (outF && ng < nvalid) outF[(size_t)m * nvalid + ng] = v;
  }
}

// ===========================================================================
extern "C" void kernel_launch(void* const* d_in, const int* in_sizes, int n_in,
                              void* d_out, int out_size, void* d_ws, size_t ws_size,
                              hipStream_t stream) {
  (void)in_sizes; (void)n_in; (void)out_size; (void)ws_size;
  const int*   charseqs = (const int*)d_in[0];
  const int*   clens    = (const int*)d_in[1];
  const int*   cids     = (const int*)d_in[2];
  const int*   wids     = (const int*)d_in[3];
  const int*   slens    = (const int*)d_in[4];
  const float* cemb     = (const float*)d_in[5];
  const float* wemb     = (const float*)d_in[6];
  const float* cWx[2]   = {(const float*)d_in[7],  (const float*)d_in[13]};
  const float* cWh[2]   = {(const float*)d_in[8],  (const float*)d_in[14]};
  const float* cbias[2] = {(const float*)d_in[9],  (const float*)d_in[15]};
  const float* sWx[2]   = {(const float*)d_in[10], (const float*)d_in[16]};
  const float* sWh[2]   = {(const float*)d_in[11], (const float*)d_in[17]};
  const float* sbias[2] = {(const float*)d_in[12], (const float*)d_in[18]};
  const float* fW1 = (const float*)d_in[19]; const float* fb1 = (const float*)d_in[20];
  const float* fW2 = (const float*)d_in[21]; const float* fb2 = (const float*)d_in[22];
  const float* fW3 = (const float*)d_in[23]; const float* fb3 = (const float*)d_in[24];

  char* base = (char*)d_ws;
  size_t off = 0;
  auto alloc = [&](size_t bytes) -> size_t {
    size_t o = off; off += (bytes + 255) & ~(size_t)255; return o;
  };

  u16* cWp[2]; float* cBp[2]; float* cC[2]; float* cH[2][2];
  for (int d = 0; d < 2; ++d) {
    cWp[d]   = (u16*)  (base + alloc((size_t)224 * 512 * 2));
    cBp[d]   = (float*)(base + alloc(448 * 4));
    cC[d]    = (float*)(base + alloc((size_t)16384 * CHP * 4));
    cH[d][0] = (float*)(base + alloc((size_t)16384 * CHP * 4));
    cH[d][1] = (float*)(base + alloc((size_t)16384 * CHP * 4));
  }
  u16* inp = (u16*)(base + alloc((size_t)256 * 128 * 512 * 2));
  u16* sOut[2] = {(u16*)(base + alloc((size_t)256 * 128 * 512 * 2)),
                  (u16*)(base + alloc((size_t)256 * 128 * 512 * 2))};
  u16* sWp[4]; float* sBp[4]; float* sC[4]; float* sH[4][2];
  for (int ld = 0; ld < 4; ++ld) {   // 0=fw L0, 1=fw L1, 2=bw L0, 3=bw L1
    sWp[ld]   = (u16*)  (base + alloc((size_t)32 * 128 * 512 * 2));
    sBp[ld]   = (float*)(base + alloc(2048 * 4));
    sC[ld]    = (float*)(base + alloc((size_t)256 * 512 * 4));
    sH[ld][0] = (float*)(base + alloc((size_t)256 * 512 * 4));
    sH[ld][1] = (float*)(base + alloc((size_t)256 * 512 * 4));
  }
  u16* sbuf = (u16*)(base + alloc((size_t)256 * 4096 * 2));
  u16* fW1p = (u16*)(base + alloc((size_t)128 * 16 * 512 * 2));
  u16* fW2p = (u16*)(base + alloc((size_t)8 * 16 * 512 * 2));
  u16* fW3p = (u16*)(base + alloc((size_t)8 * 1 * 512 * 2));
  u16* x1   = (u16*)(base + alloc((size_t)256 * 256 * 2));
  u16* x2   = (u16*)(base + alloc((size_t)256 * 256 * 2));

  // ---- pack weights & biases (L2 resident thereafter) ----
  for (int d = 0; d < 2; ++d) {
    int tot = 8 * 28 * 512;
    k_pack_lstm_w<<<(tot + 255) / 256, 256, 0, stream>>>(cWx[d], cWh[d], cWp[d],
        8, 128, 100, 100, 100, CHP, tot);
    k_pack_lstm_b<<<2, 256, 0, stream>>>(cbias[d], cBp[d], 100, CHP);
  }
  for (int ld = 0; ld < 4; ++ld) {
    int dir = ld >> 1, layer = ld & 1;
    const float* wx = sWx[dir]   + (size_t)layer * 500 * 2000;
    const float* wh = sWh[dir]   + (size_t)layer * 500 * 2000;
    const float* bb = sbias[dir] + (size_t)layer * 2000;
    int tot = 32 * 128 * 512;
    k_pack_lstm_w<<<(tot + 255) / 256, 256, 0, stream>>>(wx, wh, sWp[ld],
        32, 512, 500, 500, 500, 512, tot);
    k_pack_lstm_b<<<8, 256, 0, stream>>>(bb, sBp[ld], 500, 512);
  }
  k_pack_head_w<<<(128 * 16 * 512 + 255) / 256, 256, 0, stream>>>(fW1, fW1p, 128, 4000, 256, 128 * 16 * 512);
  k_pack_head_w<<<(8 * 16 * 512 + 255) / 256, 256, 0, stream>>>(fW2, fW2p, 8, 256, 256, 8 * 16 * 512);
  k_pack_head_w<<<(8 * 1 * 512 + 255) / 256, 256, 0, stream>>>(fW3, fW3p, 8, 256, 2, 8 * 1 * 512);

  // ---- zero-init states (workspace is poisoned each run) ----
  for (int d = 0; d < 2; ++d) {
    int n = 16384 * CHP;
    k_zero_f32<<<(n + 255) / 256, 256, 0, stream>>>(cC[d], n);
    k_zero_f32<<<(n + 255) / 256, 256, 0, stream>>>(cH[d][0], n);
  }
  for (int ld = 0; ld < 4; ++ld) {
    int n = 256 * 512;
    k_zero_f32<<<(n + 255) / 256, 256, 0, stream>>>(sC[ld], n);
    k_zero_f32<<<(n + 255) / 256, 256, 0, stream>>>(sH[ld][0], n);
  }

  // ---- char BiLSTM: 16 steps each direction (d==1 reads reversed) ----
  dim3 cgrid(1024, 7), blk(128);
  for (int t = 0; t < 16; ++t)
    for (int d = 0; d < 2; ++d)
      k_char_step<<<cgrid, blk, 0, stream>>>(charseqs, clens, cemb,
          cH[d][t & 1], cH[d][(t + 1) & 1], cC[d], cWp[d], cBp[d], t, d);
  // final h in cH[d][0] (16 steps), final c in cC[d]

  // ---- sentence input: [cf|hf|cb|hb|word_emb] -> bf16 [B,S,512] ----
  {
    int n = 256 * 128 * 512;
    k_gather_inp<<<(n + 255) / 256, 256, 0, stream>>>(cids, wids,
        cC[0], cH[0][0], cC[1], cH[1][0], wemb, inp);
  }

  // ---- sentence stacked BiLSTM: L0 before L1 within each direction ----
  struct Run { int ld; const u16* x; u16* out; int rev; };
  Run runs[4] = {{0, inp, sOut[0], 0}, {2, inp, sOut[1], 1},
                 {1, sOut[0], nullptr, 0}, {3, sOut[1], nullptr, 0}};
  dim3 sgrid(16, 32);
  for (int r = 0; r < 4; ++r) {
    int ld = runs[r].ld;
    for (int t = 0; t < 128; ++t)
      k_sent_step<<<sgrid, blk, 0, stream>>>(runs[r].x, runs[r].out,
          sH[ld][t & 1], sH[ld][(t + 1) & 1], sC[ld], sWp[ld], sBp[ld],
          slens, t, runs[r].rev);
  }
  // final h in sH[ld][0] (128 steps), final c in sC[ld]

  // ---- dense head ----
  {
    int n = 256 * 4096;
    k_pack_s<<<(n + 255) / 256, 256, 0, stream>>>(sbuf,
        sC[0], sC[1], sC[2], sC[3], sH[0][0], sH[1][0], sH[2][0], sH[3][0]);
  }
  k_head_gemm<<<dim3(16, 4), blk, 0, stream>>>(sbuf, 4096, 128, fW1p, fb1, 256, 1, x1, nullptr, 16, 256);
  k_head_gemm<<<dim3(16, 4), blk, 0, stream>>>(x1,   256,  8,  fW2p, fb2, 256, 1, x2, nullptr, 16, 256);
  k_head_gemm<<<dim3(16, 1), blk, 0, stream>>>(x2,   256,  8,  fW3p, fb3, 2,   0, nullptr, (float*)d_out, 1, 16);
}
